// GatedTransformerXLLayer_28621662060623
// MI455X (gfx1250) — matrix operand order repaired and checked by reading, not compiled, not run on hardware
//
#include <hip/hip_runtime.h>
#include <hip/hip_bf16.h>
#include <stdint.h>

// ---------------------------------------------------------------- constants
#define DD     1024
#define HH     16
#define HD     64
#define HIDN   4096
#define CURS   512
#define PREVS  512
#define FULLS  1024
#define BSZ    16
#define MROWS  (CURS*BSZ)    // 8192
#define FROWS  (FULLS*BSZ)   // 16384
#define SCALEF 0.125f

typedef _Float16 half_t;
typedef half_t v16h __attribute__((ext_vector_type(16)));
typedef float  v8f  __attribute__((ext_vector_type(8)));

union AF { v16h v; uint32_t u[8]; uint4 q4[2]; };

__device__ __forceinline__ v8f wmma16(v16h a, v16h b, v8f c) {
  return __builtin_amdgcn_wmma_f32_16x16x32_f16(false, a, false, b, (short)0, c, false, false);
}
// 16-bit A-matrix 16x32 K pattern (contiguous): lanes 0-15 get K=2v,
// lanes 16-31 get +16.
__device__ __forceinline__ int kpatf(int vv, int hf) {
  return 2*vv + 16*hf;
}
__device__ __forceinline__ v8f vzero8() {
  v8f z;
#pragma unroll
  for (int i = 0; i < 8; ++i) z[i] = 0.f;
  return z;
}
__device__ __forceinline__ uint32_t pack2h(float a, float b) {
  union { half_t h[2]; uint32_t u; } t;
  t.h[0] = (half_t)a; t.h[1] = (half_t)b; return t.u;
}

// ------------------------------------------------- CDNA5 async-to-LDS probe
#if defined(__has_builtin)
#if __has_builtin(__builtin_amdgcn_global_load_async_to_lds_b128)
#define USE_ASYNC_LDS 1
#endif
#endif
#ifndef USE_ASYNC_LDS
#define USE_ASYNC_LDS 0
#endif

// builtin wants int-vector(16B) pointees: global (AS1) src, LDS (AS3) dst
typedef int v4i_gcc __attribute__((vector_size(16)));
#define AS1(p) ((__attribute__((address_space(1))) v4i_gcc*)(void*)(p))
#define AS3(p) ((__attribute__((address_space(3))) v4i_gcc*)(void*)(p))

#if defined(__has_builtin)
#if __has_builtin(__builtin_amdgcn_s_wait_asynccnt)
#define WAIT_ASYNC(n) __builtin_amdgcn_s_wait_asynccnt((unsigned short)(n))
#endif
#endif
#ifndef WAIT_ASYNC
#define WAIT_ASYNC(n) asm volatile("s_wait_asynccnt %0" ::"n"(n))
#endif

// ---------------------------------------------------------------- converts
__global__ void __launch_bounds__(256) k_cvt(const float* __restrict__ s,
                                             half_t* __restrict__ d, int n) {
  for (int i = blockIdx.x*256 + threadIdx.x; i < n; i += gridDim.x*256)
    d[i] = (half_t)s[i];
}

// src (K x N) f32 -> dst (N x K) f16, 32x32 LDS tiles
__global__ void __launch_bounds__(256) k_cvt_t(const float* __restrict__ s,
                                               half_t* __restrict__ d, int K, int N) {
  __shared__ float tile[32][33];
  int tx = threadIdx.x, ty = threadIdx.y;          // (32, 8)
  int k0 = blockIdx.y*32, n0 = blockIdx.x*32;
#pragma unroll
  for (int r = 0; r < 4; ++r) {
    int row = ty + 8*r;
    tile[row][tx] = s[(size_t)(k0+row)*N + n0 + tx];
  }
  __syncthreads();
#pragma unroll
  for (int r = 0; r < 4; ++r) {
    int rowN = ty + 8*r;
    d[(size_t)(n0+rowN)*K + k0 + tx] = (half_t)tile[tx][rowN];
  }
}

// ---------------------------------------------------------------- layernorm
__global__ void __launch_bounds__(256) k_ln(const float* __restrict__ srcA,
                                            const float* __restrict__ srcB, int rowsA,
                                            const float* __restrict__ g,
                                            const float* __restrict__ bt,
                                            half_t* __restrict__ out) {
  int row = blockIdx.x;
  const float* src = (row < rowsA) ? srcA + (size_t)row*DD
                                   : srcB + (size_t)(row - rowsA)*DD;
  float x[4], s = 0.f, s2 = 0.f;
#pragma unroll
  for (int r = 0; r < 4; ++r) {
    x[r] = src[threadIdx.x + 256*r];
    s += x[r]; s2 += x[r]*x[r];
  }
#pragma unroll
  for (int off = 1; off < 32; off <<= 1) {
    s  += __shfl_xor(s,  off, 32);
    s2 += __shfl_xor(s2, off, 32);
  }
  __shared__ float sh[2][8];
  int wid = threadIdx.x >> 5;
  if ((threadIdx.x & 31) == 0) { sh[0][wid] = s; sh[1][wid] = s2; }
  __syncthreads();
  s = 0.f; s2 = 0.f;
#pragma unroll
  for (int i = 0; i < 8; ++i) { s += sh[0][i]; s2 += sh[1][i]; }
  float mu  = s * (1.f/DD);
  float var = s2 * (1.f/DD) - mu*mu;
  float inv = rsqrtf(var + 1e-5f);
#pragma unroll
  for (int r = 0; r < 4; ++r) {
    int col = threadIdx.x + 256*r;
    out[(size_t)row*DD + col] = (half_t)((x[r]-mu)*inv*g[col] + bt[col]);
  }
}

// ---------------------------------------------------------------- WMMA GEMM
// C(MxN) = A(MxK,f16) * Bt(NxK,f16)^T + bias
// Block tile 128x64, BK=32, double-buffered LDS (async global->LDS when
// available), 8 waves each computing a 32x32 patch (2x2 wmma tiles).
#define BM   128
#define BN   64
#define LDP  40    // padded LDS row (f16 elems): 80B rows, 16B aligned

__global__ void __launch_bounds__(256) k_gemm(const half_t* __restrict__ A,
                                              const half_t* __restrict__ Bt,
                                              const float* __restrict__ bias,
                                              float* __restrict__ Cf,
                                              half_t* __restrict__ Cb,
                                              int M, int N, int K,
                                              int lda, int ldb, int ldcf, int ldcb,
                                              int relu) {
  __shared__ __align__(16) half_t As[2][BM][LDP];
  __shared__ __align__(16) half_t Bs[2][BN][LDP];
  int nb = N >> 6;
  int bm = blockIdx.x / nb, bn = blockIdx.x % nb;
  int mb = bm*BM, nbase = bn*BN;
  int tid = threadIdx.x;
  int wi = tid >> 5, l = tid & 31, n16 = l & 15, hf = l >> 4;
  int wm = (wi >> 1)*32, wn = (wi & 1)*32;

  // staging roles: A = 512 x b128 chunks (2/thread), B = 256 x b128 (1/thread)
  int arow0 = tid >> 2, aseg = tid & 3;
  int arow1 = arow0 + 64;
  int brow  = tid >> 2;
  const half_t* gA0 = A  + (size_t)(mb + arow0)*lda + aseg*8;
  const half_t* gA1 = A  + (size_t)(mb + arow1)*lda + aseg*8;
  const half_t* gB  = Bt + (size_t)(nbase + brow)*ldb + aseg*8;

  v8f acc[4];
#pragma unroll
  for (int t = 0; t < 4; ++t) acc[t] = vzero8();
  int nk = K >> 5;

#define COMPUTE(bufi)                                                          \
  do {                                                                         \
    AF a0, a1, b0, b1;                                                         \
    const half_t* pa0 = &As[bufi][wm + n16][8*hf];                             \
    const half_t* pa1 = &As[bufi][wm + 16 + n16][8*hf];                        \
    const half_t* pb0 = &Bs[bufi][wn + n16][8*hf];                             \
    const half_t* pb1 = &Bs[bufi][wn + 16 + n16][8*hf];                        \
    a0.q4[0] = *(const uint4*)pa0;  a0.q4[1] = *(const uint4*)(pa0 + 16);      \
    a1.q4[0] = *(const uint4*)pa1;  a1.q4[1] = *(const uint4*)(pa1 + 16);      \
    b0.q4[0] = *(const uint4*)pb0;  b0.q4[1] = *(const uint4*)(pb0 + 16);      \
    b1.q4[0] = *(const uint4*)pb1;  b1.q4[1] = *(const uint4*)(pb1 + 16);      \
    acc[0] = wmma16(a0.v, b0.v, acc[0]);                                       \
    acc[1] = wmma16(a0.v, b1.v, acc[1]);                                       \
    acc[2] = wmma16(a1.v, b0.v, acc[2]);                                       \
    acc[3] = wmma16(a1.v, b1.v, acc[3]);                                       \
  } while (0)

#if USE_ASYNC_LDS
#define ISSUE(bufi, k0)                                                        \
  do {                                                                         \
    __builtin_amdgcn_global_load_async_to_lds_b128(                            \
        AS1(gA0 + (k0)), AS3(&As[bufi][arow0][aseg*8]), 0, 0);                 \
    __builtin_amdgcn_global_load_async_to_lds_b128(                            \
        AS1(gA1 + (k0)), AS3(&As[bufi][arow1][aseg*8]), 0, 0);                 \
    __builtin_amdgcn_global_load_async_to_lds_b128(                            \
        AS1(gB + (k0)),  AS3(&Bs[bufi][brow][aseg*8]), 0, 0);                  \
  } while (0)

  ISSUE(0, 0);
  for (int k = 0; k < nk; ++k) {
    if (k + 1 < nk) { ISSUE((k + 1) & 1, (k + 1) * 32); WAIT_ASYNC(3); }
    else            { WAIT_ASYNC(0); }
    __syncthreads();           // all waves' stage-k LDS writes visible
    COMPUTE(k & 1);
    __syncthreads();           // reads done before buffer is reused
  }
#else
  uint4 ra0, ra1, rb;
#define LOADR(k0)                                                              \
  do {                                                                         \
    ra0 = *(const uint4*)(gA0 + (k0));                                         \
    ra1 = *(const uint4*)(gA1 + (k0));                                         \
    rb  = *(const uint4*)(gB + (k0));                                          \
  } while (0)
#define STORER(bufi)                                                           \
  do {                                                                         \
    *(uint4*)&As[bufi][arow0][aseg*8] = ra0;                                   \
    *(uint4*)&As[bufi][arow1][aseg*8] = ra1;                                   \
    *(uint4*)&Bs[bufi][brow][aseg*8]  = rb;                                    \
  } while (0)

  LOADR(0);
  STORER(0);
  __syncthreads();
  for (int k = 0; k < nk; ++k) {
    if (k + 1 < nk) LOADR((k + 1) * 32);
    COMPUTE(k & 1);
    __syncthreads();
    if (k + 1 < nk) { STORER((k + 1) & 1); __syncthreads(); }
  }
#endif

  // epilogue
#pragma unroll
  for (int mi = 0; mi < 2; ++mi)
#pragma unroll
    for (int ni = 0; ni < 2; ++ni) {
      int col = nbase + wn + ni*16 + n16;
      float bv = bias ? bias[col] : 0.f;
      v8f av = acc[mi*2 + ni];
#pragma unroll
      for (int vv = 0; vv < 8; ++vv) {
        int row = mb + wm + mi*16 + vv + 8*hf;
        float x = av[vv] + bv;
        if (relu) x = fmaxf(x, 0.f);
        if (Cf) Cf[(size_t)row*ldcf + col] = x;
        if (Cb) Cb[(size_t)row*ldcb + col] = (half_t)x;
      }
    }
}

// ---------------------------------------------------------------- attention
// Fused flash-style Transformer-XL attention. One wave per (b, h, 16-row tile).
// rel_shift folded: position[i,j] = posraw[i, j-i+511]; 16-aligned tiles touch
// exactly two aligned 16-wide delta-tiles -> 4-slot LDS ring per wave.
__global__ void __launch_bounds__(128) k_attn(const float* __restrict__ q,
                                              const half_t* __restrict__ kv,
                                              const half_t* __restrict__ rbf,
                                              const float* __restrict__ uvec,
                                              const float* __restrict__ vvec,
                                              half_t* __restrict__ outp) {
  __shared__ float posL[4][4][16][16];                 // [wave][slot][m][n]
  __shared__ __align__(16) half_t pbuf[4][16][32];     // P re-layout C->A frag
  int wi = threadIdx.x >> 5, l = threadIdx.x & 31;
  int w  = blockIdx.x*4 + wi;
  int it0 = w & 31;
  int bh  = w >> 5;
  int h = bh & 15, b = bh >> 4;
  int i0 = it0*16;
  int n16 = l & 15, hf = l >> 4;

  const float* qrow = q + (size_t)((i0 + n16)*BSZ + b)*DD + h*HD;
  const float* uh = uvec + h*HD;
  const float* vh = vvec + h*HD;
  AF au[2], av[2];
#pragma unroll
  for (int c = 0; c < 2; ++c)
#pragma unroll
    for (int vv = 0; vv < 8; ++vv) {
      int k = c*32 + kpatf(vv, hf);
      float q0 = qrow[k], q1 = qrow[k+1];
      au[c].u[vv] = pack2h(q0 + uh[k], q1 + uh[k+1]);
      av[c].u[vv] = pack2h(q0 + vh[k], q1 + vh[k+1]);
    }

  v8f O[4];
#pragma unroll
  for (int dt = 0; dt < 4; ++dt) O[dt] = vzero8();
  float rm[8], rs[8];
#pragma unroll
  for (int vv = 0; vv < 8; ++vv) { rm[vv] = -1e30f; rs[vv] = 0.f; }

  int ntile = 31 - it0;
  int jjend = (it0 + 32) >> 1;        // causal: j <= i+512
  for (int jj = 0; jj <= jjend; ++jj) {
    int hi = 2*jj + 33 - it0;
    for (; ntile <= hi; ++ntile) {    // produce pos delta-tiles into ring
      int dr = 16*ntile + n16; if (dr > FULLS-1) dr = FULLS-1;
      const half_t* rp = rbf + (size_t)dr*DD + h*HD + 16*hf;
      AF b0, b1;
      b0.q4[0] = ((const uint4*)rp)[0];      b0.q4[1] = ((const uint4*)rp)[1];
      b1.q4[0] = ((const uint4*)(rp+32))[0]; b1.q4[1] = ((const uint4*)(rp+32))[1];
      v8f pa = vzero8();
      pa = wmma16(av[0].v, b0.v, pa);
      pa = wmma16(av[1].v, b1.v, pa);
      int slot = ntile & 3;
#pragma unroll
      for (int vv = 0; vv < 8; ++vv) posL[wi][slot][vv + 8*hf][n16] = pa[vv];
    }
    v8f ct[2];
#pragma unroll
    for (int t = 0; t < 2; ++t) {
      int j0t = jj*32 + t*16;
      const half_t* kp = kv + (size_t)((j0t + n16)*BSZ + b)*(2*DD) + h*HD + 16*hf;
      AF b0, b1;
      b0.q4[0] = ((const uint4*)kp)[0];      b0.q4[1] = ((const uint4*)kp)[1];
      b1.q4[0] = ((const uint4*)(kp+32))[0]; b1.q4[1] = ((const uint4*)(kp+32))[1];
      v8f c = vzero8();
      c = wmma16(au[0].v, b0.v, c);
      c = wmma16(au[1].v, b1.v, c);
      ct[t] = c;
    }
    float s[2][8];
#pragma unroll
    for (int t = 0; t < 2; ++t)
#pragma unroll
      for (int vv = 0; vv < 8; ++vv) {
        int mi = vv + 8*hf;
        int j  = jj*32 + t*16 + n16;
        int c  = 15 + n16 - mi;                        // 0..30
        int T  = 2*jj + t + 31 - it0 + (c >> 4);
        float pos = posL[wi][T & 3][mi][c & 15];
        float val = (ct[t][vv] + pos) * SCALEF;
        s[t][vv] = (j > i0 + mi + PREVS) ? -1e30f : val;
      }
    float tm[8];
#pragma unroll
    for (int vv = 0; vv < 8; ++vv) tm[vv] = fmaxf(s[0][vv], s[1][vv]);
#pragma unroll
    for (int off = 1; off < 16; off <<= 1)
#pragma unroll
      for (int vv = 0; vv < 8; ++vv) tm[vv] = fmaxf(tm[vv], __shfl_xor(tm[vv], off, 32));
    float cf[8];
#pragma unroll
    for (int vv = 0; vv < 8; ++vv) {
      float nm = fmaxf(rm[vv], tm[vv]);
      cf[vv] = __expf(rm[vv] - nm);
      rm[vv] = nm;
    }
    float ps[8];
#pragma unroll
    for (int vv = 0; vv < 8; ++vv) {
      s[0][vv] = __expf(s[0][vv] - rm[vv]);
      s[1][vv] = __expf(s[1][vv] - rm[vv]);
      ps[vv] = s[0][vv] + s[1][vv];
    }
#pragma unroll
    for (int off = 1; off < 16; off <<= 1)
#pragma unroll
      for (int vv = 0; vv < 8; ++vv) ps[vv] += __shfl_xor(ps[vv], off, 32);
#pragma unroll
    for (int vv = 0; vv < 8; ++vv) {
      rs[vv] = rs[vv]*cf[vv] + ps[vv];
#pragma unroll
      for (int dt = 0; dt < 4; ++dt) O[dt][vv] *= cf[vv];
    }
#pragma unroll
    for (int t = 0; t < 2; ++t)
#pragma unroll
      for (int vv = 0; vv < 8; ++vv)
        pbuf[wi][vv + 8*hf][t*16 + n16] = (half_t)s[t][vv];
    __builtin_amdgcn_wave_barrier();
    AF pf;
#pragma unroll
    for (int vv = 0; vv < 8; ++vv)
      pf.u[vv] = *(const uint32_t*)&pbuf[wi][n16][kpatf(vv, hf)];
#pragma unroll
    for (int dt = 0; dt < 4; ++dt) {
      AF bvf;
#pragma unroll
      for (int vv = 0; vv < 8; ++vv) {
        int k = 2*vv + 16*hf;
        const half_t* vp = kv + (size_t)((jj*32 + k)*BSZ + b)*(2*DD)
                              + DD + h*HD + dt*16 + n16;
        union { half_t h2[2]; uint32_t u; } pk;
        pk.h2[0] = vp[0];
        pk.h2[1] = vp[(size_t)BSZ*2*DD];
        bvf.u[vv] = pk.u;
      }
      O[dt] = wmma16(pf.v, bvf.v, O[dt]);
    }
  }
#pragma unroll
  for (int vv = 0; vv < 8; ++vv) {
    int mi = vv + 8*hf;
    float inv = 1.f / rs[vv];
    half_t* orow = outp + (size_t)((i0 + mi)*BSZ + b)*DD + h*HD;
#pragma unroll
    for (int dt = 0; dt < 4; ++dt)
      orow[dt*16 + n16] = (half_t)(O[dt][vv] * inv);
  }
}

// ---------------------------------------------------------------- GRU gates
__global__ void __launch_bounds__(256) k_gateA(const float* __restrict__ yW,   // ld 3072
                                               const float* __restrict__ xU,   // ld 2048
                                               const float* __restrict__ x,    // ld 1024
                                               const float* __restrict__ bg,
                                               float* __restrict__ zz,
                                               half_t* __restrict__ rx, int n) {
  for (int i = blockIdx.x*256 + threadIdx.x; i < n; i += gridDim.x*256) {
    int row = i >> 10, col = i & 1023;
    float tr = yW[(size_t)row*3072 + col]        + xU[(size_t)row*2048 + col];
    float tz = yW[(size_t)row*3072 + 1024 + col] + xU[(size_t)row*2048 + 1024 + col] - bg[col];
    float r = 1.f/(1.f + __expf(-tr));
    float z = 1.f/(1.f + __expf(-tz));
    zz[i] = z;
    rx[i] = (half_t)(r * x[i]);
  }
}

__global__ void __launch_bounds__(256) k_gateB(const float* __restrict__ yW,
                                               const float* __restrict__ rxU2,
                                               const float* __restrict__ zz,
                                               const float* __restrict__ x,
                                               float* __restrict__ of,
                                               half_t* __restrict__ ob, int n) {
  for (int i = blockIdx.x*256 + threadIdx.x; i < n; i += gridDim.x*256) {
    int row = i >> 10, col = i & 1023;
    float hh = tanhf(yW[(size_t)row*3072 + 2048 + col] + rxU2[i]);
    float z  = zz[i];
    float o  = (1.f - z)*x[i] + z*hh;
    of[i] = o;
    if (ob) ob[i] = (half_t)o;
  }
}

// ---------------------------------------------------------------- host
static void launch_gemm(const half_t* A, const half_t* Bt, const float* bias,
                        float* Cf, half_t* Cb, int M, int N, int K,
                        int lda, int ldb, int ldcf, int ldcb, int relu,
                        hipStream_t s) {
  int blocks = (M/BM) * (N/BN);
  k_gemm<<<blocks, 256, 0, s>>>(A, Bt, bias, Cf, Cb, M, N, K, lda, ldb, ldcf, ldcb, relu);
}

extern "C" void kernel_launch(void* const* d_in, const int* in_sizes, int n_in,
                              void* d_out, int out_size, void* d_ws, size_t ws_size,
                              hipStream_t stream) {
  const float* inputs = (const float*)d_in[0];
  const float* pos    = (const float*)d_in[1];
  const float* uvec   = (const float*)d_in[2];
  const float* vvec   = (const float*)d_in[3];
  const float* memory = (const float*)d_in[4];
  const float* Wkv = (const float*)d_in[6];  const float* bkv = (const float*)d_in[7];
  const float* Wq  = (const float*)d_in[8];  const float* bq  = (const float*)d_in[9];
  const float* Wo  = (const float*)d_in[10]; const float* bo  = (const float*)d_in[11];
  const float* Wp  = (const float*)d_in[12]; const float* bp  = (const float*)d_in[13];
  const float* ln1g = (const float*)d_in[14]; const float* ln1b = (const float*)d_in[15];
  const float* ln2g = (const float*)d_in[16]; const float* ln2b = (const float*)d_in[17];
  const float* W1 = (const float*)d_in[18]; const float* b1 = (const float*)d_in[19];
  const float* W2 = (const float*)d_in[20]; const float* b2 = (const float*)d_in[21];
  const float* g1W = (const float*)d_in[22]; const float* g1U = (const float*)d_in[23];
  const float* g1bg = (const float*)d_in[24];
  const float* g2W = (const float*)d_in[25]; const float* g2U = (const float*)d_in[26];
  const float* g2bg = (const float*)d_in[27];
  float* out = (float*)d_out;

  char* ws = (char*)d_ws;
  size_t cur = 0;
  auto alloc = [&](size_t bytes) -> char* {
    char* p = ws + cur;
    cur = (cur + bytes + 255) & ~(size_t)255;
    return p;
  };
  half_t* Wkvt = (half_t*)alloc((size_t)2048*1024*2);
  half_t* Wqt  = (half_t*)alloc((size_t)1024*1024*2);
  half_t* Wpt  = (half_t*)alloc((size_t)1024*1024*2);
  half_t* Wot  = (half_t*)alloc((size_t)1024*1024*2);
  half_t* W1t  = (half_t*)alloc((size_t)4096*1024*2);
  half_t* W2t  = (half_t*)alloc((size_t)1024*4096*2);
  half_t* g1Wb = (half_t*)alloc((size_t)3*1024*1024*2);
  half_t* g1Ub = (half_t*)alloc((size_t)3*1024*1024*2);
  half_t* g2Wb = (half_t*)alloc((size_t)3*1024*1024*2);
  half_t* g2Ub = (half_t*)alloc((size_t)3*1024*1024*2);
  half_t* xbf  = (half_t*)alloc((size_t)MROWS*DD*2);
  half_t* posb = (half_t*)alloc((size_t)FULLS*DD*2);
  half_t* x1   = (half_t*)alloc((size_t)FROWS*DD*2);      // overlay: zz (f32, same bytes)
  half_t* kvb  = (half_t*)alloc((size_t)FROWS*2*DD*2);    // kv + q contiguous = yW overlay
  float*  qf   = (float*) alloc((size_t)MROWS*DD*4);
  half_t* rbf  = (half_t*)alloc((size_t)FULLS*DD*2);
  half_t* att  = (half_t*)alloc((size_t)MROWS*DD*2);      // overlay: rx
  half_t* a1   = (half_t*)alloc((size_t)MROWS*DD*2);      // overlay: m2
  float*  xU   = (float*) alloc((size_t)MROWS*2048*4);    // overlay: t1 (f16, same bytes)
  float*  rxU2 = (float*) alloc((size_t)MROWS*DD*4);
  float*  o1f  = (float*) alloc((size_t)MROWS*DD*4);
  half_t* o1b  = (half_t*)alloc((size_t)MROWS*DD*2);
  half_t* x2   = (half_t*)alloc((size_t)MROWS*DD*2);
  if (cur > ws_size) return;   // workspace too small; bail deterministically

  float*  yW  = (float*)kvb;                // 8192x3072 f32 == kv+q bytes exactly
  float*  zz  = (float*)x1;
  half_t* rx  = att;
  half_t* t1  = (half_t*)xU;
  half_t* m2  = a1;

  dim3 t32x8(32, 8);
  k_cvt_t<<<dim3(64, 32),  t32x8, 0, stream>>>(Wkv, Wkvt, 1024, 2048);
  k_cvt_t<<<dim3(32, 32),  t32x8, 0, stream>>>(Wq,  Wqt,  1024, 1024);
  k_cvt_t<<<dim3(32, 32),  t32x8, 0, stream>>>(Wp,  Wpt,  1024, 1024);
  k_cvt_t<<<dim3(32, 32),  t32x8, 0, stream>>>(Wo,  Wot,  1024, 1024);
  k_cvt_t<<<dim3(128, 32), t32x8, 0, stream>>>(W1,  W1t,  1024, 4096);
  k_cvt_t<<<dim3(32, 128), t32x8, 0, stream>>>(W2,  W2t,  4096, 1024);
  k_cvt<<<4096, 256, 0, stream>>>(g1W, g1Wb, 3*1024*1024);
  k_cvt<<<4096, 256, 0, stream>>>(g1U, g1Ub, 3*1024*1024);
  k_cvt<<<4096, 256, 0, stream>>>(g2W, g2Wb, 3*1024*1024);
  k_cvt<<<4096, 256, 0, stream>>>(g2U, g2Ub, 3*1024*1024);
  k_cvt<<<4096, 256, 0, stream>>>(inputs, xbf, MROWS*DD);
  k_cvt<<<4096, 256, 0, stream>>>(pos, posb, FULLS*DD);

  // x1 = LN1(concat(memory, inputs))
  k_ln<<<FROWS, 256, 0, stream>>>(memory, inputs, PREVS*BSZ, ln1g, ln1b, x1);
  // kv = x1 @ Wkv + bkv  (f16 out)
  launch_gemm(x1, Wkvt, bkv, nullptr, kvb, FROWS, 2048, 1024, 1024, 1024, 0, 2048, 0, stream);
  // q  = x1[PREV:] @ Wq + bq  (f32 out)
  launch_gemm(x1 + (size_t)PREVS*BSZ*DD, Wqt, bq, qf, nullptr, MROWS, 1024, 1024,
              1024, 1024, 1024, 0, 0, stream);
  // r  = pos @ Wp + bp  (f16 out)
  launch_gemm(posb, Wpt, bp, nullptr, rbf, FULLS, 1024, 1024, 1024, 1024, 0, 1024, 0, stream);
  // fused flash attention (rel_shift + mask + softmax + P@V)
  k_attn<<<2048, 128, 0, stream>>>(qf, kvb, rbf, uvec, vvec, att);
  // a1 = relu(att @ Wo + bo)
  launch_gemm(att, Wot, bo, nullptr, a1, MROWS, 1024, 1024, 1024, 1024, 0, 1024, 1, stream);

  // ---- GRU gate 1: x = inputs, y = a1
  launch_gemm(a1,  g1Wb, nullptr, yW, nullptr, MROWS, 3072, 1024, 1024, 1024, 3072, 0, 0, stream);
  launch_gemm(xbf, g1Ub, nullptr, xU, nullptr, MROWS, 2048, 1024, 1024, 1024, 2048, 0, 0, stream);
  k_gateA<<<8192, 256, 0, stream>>>(yW, xU, inputs, g1bg, zz, rx, MROWS*DD);
  launch_gemm(rx, g1Ub + (size_t)2*1024*1024, nullptr, rxU2, nullptr,
              MROWS, 1024, 1024, 1024, 1024, 1024, 0, 0, stream);
  k_gateB<<<8192, 256, 0, stream>>>(yW, rxU2, zz, inputs, o1f, o1b, MROWS*DD);

  // MLP
  k_ln<<<MROWS, 256, 0, stream>>>(o1f, nullptr, MROWS, ln2g, ln2b, x2);
  launch_gemm(x2, W1t, b1, nullptr, t1, MROWS, 4096, 1024, 1024, 1024, 0, 4096, 1, stream);
  launch_gemm(t1, W2t, b2, nullptr, m2, MROWS, 1024, 4096, 4096, 4096, 0, 1024, 1, stream);

  // ---- GRU gate 2: x = o1, y = m2 ; output -> d_out
  launch_gemm(m2,  g2Wb, nullptr, yW, nullptr, MROWS, 3072, 1024, 1024, 1024, 3072, 0, 0, stream);
  launch_gemm(o1b, g2Ub, nullptr, xU, nullptr, MROWS, 2048, 1024, 1024, 1024, 2048, 0, 0, stream);
  k_gateA<<<8192, 256, 0, stream>>>(yW, xU, o1f, g2bg, zz, rx, MROWS*DD);
  launch_gemm(rx, g2Ub + (size_t)2*1024*1024, nullptr, rxU2, nullptr,
              MROWS, 1024, 1024, 1024, 1024, 1024, 0, 0, stream);
  k_gateB<<<8192, 256, 0, stream>>>(yW, rxU2, zz, o1f, out, nullptr, MROWS*DD);
}
